// EdgeAwareBlock_10668698764068
// MI455X (gfx1250) — matrix operand order repaired and checked by reading, not compiled, not run on hardware
//
#include <hip/hip_runtime.h>
#include <hip/hip_bf16.h>

#define IN_DIM 256
#define HID 256
#define EDGE_DIM 64

typedef __attribute__((ext_vector_type(16))) _Float16 v16h;
typedef __attribute__((ext_vector_type(8)))  float    v8f;

union Frag16 { v16h v; uint4 q[2]; };

__device__ __forceinline__ unsigned short f2h_bits(float x) {
  union { _Float16 h; unsigned short u; } c;
  c.h = (_Float16)x;
  return c.u;
}

// dst[n*K + k] = (f16) src[k*N + n]   (column-major f16 copy for WMMA B-frags)
__global__ void k_transpose_f16(const float* __restrict__ src,
                                unsigned short* __restrict__ dst,
                                int K, int N) {
  int k = blockIdx.x;
  int n = threadIdx.x;
  if (k < K && n < N) dst[n * K + k] = f2h_bits(src[(size_t)k * N + n]);
}

// -------- Edge message + scatter: aggr[dst] += relu(x[src] + edge_attr@We + be)
// 256 threads = 8 waves; 32 edges per wave; 256 edges per block.
__global__ __launch_bounds__(256)
void k_edge_scatter(const float* __restrict__ x,
                    const int* __restrict__ ei,             // [2][E]
                    const float* __restrict__ edge_attr,    // [E][64]
                    const unsigned short* __restrict__ WeT, // [256][64] f16 col-major
                    const float* __restrict__ be,           // [256]
                    float* __restrict__ aggr,               // [N+1][256]; row N = OOB dump
                    int E, int Nn) {
  __shared__ unsigned short sWeT[IN_DIM * EDGE_DIM]; // 32KB
  __shared__ unsigned short sA[8][32 * EDGE_DIM];    // 32KB, per-wave 32x64 f16 tile
  __shared__ float sBe[IN_DIM];
  __shared__ int sSrc[256];
  __shared__ int sDst[256];

  const int t    = threadIdx.x;
  const int wave = t >> 5;
  const int lane = t & 31;
  const int grp  = lane >> 4; // 0 or 1
  const int ln   = lane & 15;

  for (int i = t; i < IN_DIM * EDGE_DIM; i += 256) sWeT[i] = WeT[i];
  if (t < IN_DIM) sBe[t] = be[t];

  const long long e0 = (long long)blockIdx.x * 256;
  {
    long long e = e0 + t;
    int s_ = 0, d_ = Nn;               // OOB edges: src row 0, dst -> dummy row Nn
    if (e < E) { s_ = ei[e]; d_ = ei[(long long)E + e]; }
    sSrc[t] = s_; sDst[t] = d_;
  }

  // stage this wave's 32x64 edge_attr tile as f16 in LDS
  {
    long long ebase = e0 + wave * 32;
    for (int i = lane; i < 32 * EDGE_DIM; i += 32) {
      int m = i >> 6;
      int k = i & 63;
      long long e = ebase + m;
      float v = (e < E) ? edge_attr[e * EDGE_DIM + k] : 0.0f;
      sA[wave][m * EDGE_DIM + k] = f2h_bits(v);
    }
  }
  __syncthreads();

  // A fragments: two M-tiles (16 edges each), two K-steps of 32
  // layout: lane m=ln; halves0-7: K=grp*8+h ; halves8-15: K=16+grp*8+h
  Frag16 a[2][2];
#pragma unroll
  for (int mt = 0; mt < 2; ++mt) {
    const unsigned short* base = &sA[wave][(mt * 16 + ln) * EDGE_DIM];
#pragma unroll
    for (int s = 0; s < 2; ++s) {
      a[mt][s].q[0] = *(const uint4*)(base + s * 32 + grp * 8);
      a[mt][s].q[1] = *(const uint4*)(base + s * 32 + 16 + grp * 8);
    }
  }

  // 32-bit element offsets for gather/scatter rows -> SADDR-form loads/atomics
  // (C layout: VGPR r -> M = r + 8*grp, N = ln)
  unsigned xoff[16], aoff[16];
#pragma unroll
  for (int i = 0; i < 16; ++i) {
    int mt = i >> 3, r = i & 7;
    int m  = mt * 16 + r + grp * 8;    // edge within wave's 32
    xoff[i] = (unsigned)sSrc[wave * 32 + m] * IN_DIM;
    aoff[i] = (unsigned)sDst[wave * 32 + m] * IN_DIM;
  }

  for (int tt = 0; tt < 16; ++tt) {
    const int col = tt * 16 + ln;

    // Batch-issue all 16 x gathers for this tile up front (overlap with WMMAs)
    float xv[16];
#pragma unroll
    for (int i = 0; i < 16; ++i) xv[i] = x[(size_t)(xoff[i] + (unsigned)col)];

    // B fragments: lane n=ln; halves0-7: K = s*32 + grp*8 + h ; halves8-15: K = s*32 + 16 + grp*8 + h
    Frag16 b0, b1;
    const unsigned short* wb = &sWeT[col * EDGE_DIM];
    b0.q[0] = *(const uint4*)(wb + 0  + grp * 8);
    b0.q[1] = *(const uint4*)(wb + 16 + grp * 8);
    b1.q[0] = *(const uint4*)(wb + 32 + grp * 8);
    b1.q[1] = *(const uint4*)(wb + 48 + grp * 8);

    v8f acc0 = {}, acc1 = {};
    acc0 = __builtin_amdgcn_wmma_f32_16x16x32_f16(false, a[0][0].v, false, b0.v, (short)0, acc0, false, false);
    acc0 = __builtin_amdgcn_wmma_f32_16x16x32_f16(false, a[0][1].v, false, b1.v, (short)0, acc0, false, false);
    acc1 = __builtin_amdgcn_wmma_f32_16x16x32_f16(false, a[1][0].v, false, b0.v, (short)0, acc1, false, false);
    acc1 = __builtin_amdgcn_wmma_f32_16x16x32_f16(false, a[1][1].v, false, b1.v, (short)0, acc1, false, false);

    const float bev = sBe[col];

    float v0[8], v1[8];
#pragma unroll
    for (int r = 0; r < 8; ++r) {
      float v = acc0[r] + xv[r] + bev;
      v0[r] = v > 0.0f ? v : 0.0f;
    }
#pragma unroll
    for (int r = 0; r < 8; ++r) {
      float v = acc1[r] + xv[8 + r] + bev;
      v1[r] = v > 0.0f ? v : 0.0f;
    }
#pragma unroll
    for (int r = 0; r < 8; ++r)
      unsafeAtomicAdd(&aggr[(size_t)(aoff[r] + (unsigned)col)], v0[r]);
#pragma unroll
    for (int r = 0; r < 8; ++r)
      unsafeAtomicAdd(&aggr[(size_t)(aoff[8 + r] + (unsigned)col)], v1[r]);
  }
}

// -------- Node update: out = LN(relu(relu((x+aggr)@W1+b1)@W2+b2) + x)
__global__ __launch_bounds__(256)
void k_node_update(const float* __restrict__ x,
                   const float* __restrict__ aggr,
                   const unsigned short* __restrict__ W1T, // [256][256] f16 col-major
                   const float* __restrict__ b1,
                   const unsigned short* __restrict__ W2T, // [256][256] f16 col-major
                   const float* __restrict__ b2,
                   const float* __restrict__ gamma,
                   const float* __restrict__ beta,
                   float* __restrict__ out,
                   int Nn) {
  __shared__ unsigned short sZ[16 * HID];   // 8KB
  __shared__ unsigned short sH1[16 * HID];  // 8KB
  __shared__ float sH[16 * HID];            // 16KB
  __shared__ float sPartSum[16][16];
  __shared__ float sPartSq[16][16];
  __shared__ float sMu[16], sRs[16];

  const int t    = threadIdx.x;
  const int wave = t >> 5;
  const int lane = t & 31;
  const int grp  = lane >> 4;
  const int ln   = lane & 15;
  const long long r0 = (long long)blockIdx.x * 16;

  // z = x + aggr (f16 in LDS, row-major [m][k]); clamp row for tail safety
  for (int i = t; i < 16 * HID; i += 256) {
    int m = i >> 8;
    int c = i & 255;
    long long row = r0 + m;
    if (row >= Nn) row = Nn - 1;
    float v = x[row * HID + c] + aggr[row * HID + c];
    sZ[i] = f2h_bits(v);
  }
  __syncthreads();

  // h1 = relu(z @ W1 + b1) -> LDS f16
#pragma unroll
  for (int tv = 0; tv < 2; ++tv) {
    const int tt = wave + tv * 8;
    v8f acc = {};
#pragma unroll
    for (int kc = 0; kc < 8; ++kc) {
      Frag16 a, b;
      const unsigned short* za = &sZ[ln * HID + kc * 32];
      a.q[0] = *(const uint4*)(za + grp * 8);
      a.q[1] = *(const uint4*)(za + 16 + grp * 8);
      const unsigned short* wb = &W1T[(size_t)(tt * 16 + ln) * HID + kc * 32 + grp * 8];
      b.q[0] = *(const uint4*)(wb);
      b.q[1] = *(const uint4*)(wb + 16);
      acc = __builtin_amdgcn_wmma_f32_16x16x32_f16(false, a.v, false, b.v, (short)0, acc, false, false);
    }
    const int col = tt * 16 + ln;
    const float bv = b1[col];
#pragma unroll
    for (int r = 0; r < 8; ++r) {
      int m = r + grp * 8;
      float v = acc[r] + bv;
      v = v > 0.0f ? v : 0.0f;
      sH1[m * HID + col] = f2h_bits(v);
    }
  }
  __syncthreads();

  // h = relu(h1 @ W2 + b2) + x  -> LDS f32
#pragma unroll
  for (int tv = 0; tv < 2; ++tv) {
    const int tt = wave + tv * 8;
    v8f acc = {};
#pragma unroll
    for (int kc = 0; kc < 8; ++kc) {
      Frag16 a, b;
      const unsigned short* za = &sH1[ln * HID + kc * 32];
      a.q[0] = *(const uint4*)(za + grp * 8);
      a.q[1] = *(const uint4*)(za + 16 + grp * 8);
      const unsigned short* wb = &W2T[(size_t)(tt * 16 + ln) * HID + kc * 32 + grp * 8];
      b.q[0] = *(const uint4*)(wb);
      b.q[1] = *(const uint4*)(wb + 16);
      acc = __builtin_amdgcn_wmma_f32_16x16x32_f16(false, a.v, false, b.v, (short)0, acc, false, false);
    }
    const int col = tt * 16 + ln;
    const float bv = b2[col];
#pragma unroll
    for (int r = 0; r < 8; ++r) {
      int m = r + grp * 8;
      long long row = r0 + m;
      if (row >= Nn) row = Nn - 1;
      float v = acc[r] + bv;
      v = v > 0.0f ? v : 0.0f;
      sH[m * HID + col] = v + x[row * HID + col];
    }
  }
  __syncthreads();

  // LayerNorm over each of the 16 rows
  {
    int row = t >> 4;
    int sub = t & 15;
    float s = 0.0f, ss = 0.0f;
#pragma unroll
    for (int i = 0; i < 16; ++i) {
      float v = sH[row * HID + sub * 16 + i];
      s += v; ss += v * v;
    }
    sPartSum[row][sub] = s;
    sPartSq[row][sub] = ss;
  }
  __syncthreads();
  if (t < 16) {
    float s = 0.0f, ss = 0.0f;
    for (int i = 0; i < 16; ++i) { s += sPartSum[t][i]; ss += sPartSq[t][i]; }
    float mu = s * (1.0f / HID);
    float var = ss * (1.0f / HID) - mu * mu;
    sMu[t] = mu;
    sRs[t] = rsqrtf(var + 1e-5f);
  }
  __syncthreads();
  for (int i = t; i < 16 * HID; i += 256) {
    int m = i >> 8;
    int c = i & 255;
    long long row = r0 + m;
    if (row < Nn)
      out[row * HID + c] = (sH[i] - sMu[m]) * sRs[m] * gamma[c] + beta[c];
  }
}

extern "C" void kernel_launch(void* const* d_in, const int* in_sizes, int n_in,
                              void* d_out, int out_size, void* d_ws, size_t ws_size,
                              hipStream_t stream) {
  const float* x         = (const float*)d_in[0];
  const int*   ei        = (const int*)d_in[1];
  const float* edge_attr = (const float*)d_in[2];
  const float* We        = (const float*)d_in[3];
  const float* be        = (const float*)d_in[4];
  const float* W1        = (const float*)d_in[5];
  const float* b1        = (const float*)d_in[6];
  const float* W2        = (const float*)d_in[7];
  const float* b2        = (const float*)d_in[8];
  const float* gamma     = (const float*)d_in[9];
  const float* beta      = (const float*)d_in[10];
  float* out = (float*)d_out;

  const int Nn = in_sizes[0] / IN_DIM;
  const int E  = in_sizes[2] / EDGE_DIM;

  char* ws = (char*)d_ws;
  float* aggr = (float*)ws;                       // (Nn+1) rows; last row = OOB dump
  size_t off = (size_t)(Nn + 1) * IN_DIM * sizeof(float);
  unsigned short* WeT = (unsigned short*)(ws + off); off += (size_t)IN_DIM * EDGE_DIM * 2;
  unsigned short* W1T = (unsigned short*)(ws + off); off += (size_t)HID * IN_DIM * 2;
  unsigned short* W2T = (unsigned short*)(ws + off); off += (size_t)HID * HID * 2;

  hipMemsetAsync(aggr, 0, (size_t)(Nn + 1) * IN_DIM * sizeof(float), stream);
  k_transpose_f16<<<EDGE_DIM, IN_DIM, 0, stream>>>(We, WeT, EDGE_DIM, IN_DIM);
  k_transpose_f16<<<IN_DIM, HID, 0, stream>>>(W1, W1T, IN_DIM, HID);
  k_transpose_f16<<<HID, HID, 0, stream>>>(W2, W2T, HID, HID);

  k_edge_scatter<<<(E + 255) / 256, 256, 0, stream>>>(x, ei, edge_attr, WeT, be, aggr, E, Nn);
  k_node_update<<<(Nn + 15) / 16, 256, 0, stream>>>(x, aggr, W1T, b1, W2T, b2, gamma, beta, out, Nn);
}
